// LanguageConditionedEncoder_55439437857005
// MI455X (gfx1250) — matrix operand order repaired and checked by reading, not compiled, not run on hardware
//
#include <hip/hip_runtime.h>
#include <hip/hip_bf16.h>
#include <math.h>

typedef __attribute__((ext_vector_type(16))) _Float16 v16h;
typedef __attribute__((ext_vector_type(8)))  _Float16 v8h;
typedef __attribute__((ext_vector_type(8)))  float    v8f;

#define BN_ 4
#define NS_ 256
#define NG_ 16
#define HD_ 512
#define NH_ 8
#define PF_ 63     // folded edge-feature dim (126 = 2x63)
#define PFP 64     // PF_ padded to 64 so every GEMM K/N is a multiple of 32
#define LL_ 2
#define RR_ 5

// ---------------------------------------------------------------------------
// Batched GEMM, f16 x f16 -> f32 via v_wmma_f32_16x16x32_f16.
// One wave/block computes a 16(M) x 32(N) tile (1 A frag -> 2 WMMAs / k-step).
// A[m,k] = A[m*lda + k]      (k contiguous)
// B[k,n] = B[n*sbn + k]      (k contiguous -- all weights pre-transposed)
// Interior tiles: 128-bit staging loads/stores; edges: clamped masked loads.
// Fragments are gathered from LDS as aligned v8h vectors (ds_load_b128).
// ---------------------------------------------------------------------------
__global__ __launch_bounds__(32)
void gemm_wmma(const _Float16* __restrict__ A, int lda, long long sA1, long long sA2,
               const _Float16* __restrict__ B, int sbn, long long sB1, long long sB2,
               float* __restrict__ C, int ldc, long long sC1, long long sC2,
               _Float16* __restrict__ C16, int ldc16, long long sC161, long long sC162,
               const float* __restrict__ bias,
               int M, int N, int K, int innerB, int accumulate)
{
    int z  = blockIdx.z;
    int zo = z / innerB, zi = z % innerB;
    A += zo * sA1 + zi * sA2;
    B += zo * sB1 + zi * sB2;
    long long cOff   = (long long)zo * sC1   + (long long)zi * sC2;
    long long c16Off = (long long)zo * sC161 + (long long)zi * sC162;

    int n0 = blockIdx.x * 32, m0 = blockIdx.y * 16;
    __shared__ _Float16 As[16][48];   // [m][k], 96B rows -> 16B-aligned vectors
    __shared__ _Float16 Bs[32][48];   // [n][k], k contiguous
    int t = threadIdx.x;
    v8f acc0 = {}, acc1 = {};

    const bool mfull = (m0 + 16 <= M);
    const bool nfull = (n0 + 32 <= N);
    const bool vecA  = mfull && ((lda & 7) == 0) && ((((size_t)A) & 15) == 0);
    const bool vecB  = nfull && ((sbn & 7) == 0) && ((((size_t)B) & 15) == 0);

    for (int k0 = 0; k0 < K; k0 += 32) {
        const bool kfull = (k0 + 32 <= K);
        // ---- stage A: 16(m) x 32(k) ----
        if (vecA && kfull) {
            #pragma unroll
            for (int j = 0; j < 2; ++j) {
                int idx = t + 32 * j;                 // 0..63
                int r = idx >> 2, seg = idx & 3;
                *(uint4*)&As[r][seg * 8] =
                    *(const uint4*)(A + (long long)(m0 + r) * lda + k0 + seg * 8);
            }
        } else {
            const unsigned short* A16 = (const unsigned short*)A;
            for (int i = t; i < 512; i += 32) {
                int r = i >> 5, kk = i & 31;
                int m = m0 + r, k = k0 + kk;
                int mc = m < M ? m : M - 1;
                int kc = k < K ? k : K - 1;
                unsigned short raw = A16[(long long)mc * lda + kc];
                raw &= (unsigned short)-(int)(m < M && k < K);
                *(unsigned short*)&As[r][kk] = raw;
            }
        }
        // ---- stage B: 32(n) x 32(k) ----
        if (vecB && kfull) {
            #pragma unroll
            for (int j = 0; j < 4; ++j) {
                int idx = t + 32 * j;                 // 0..127
                int c = idx >> 2, seg = idx & 3;
                *(uint4*)&Bs[c][seg * 8] =
                    *(const uint4*)(B + (long long)(n0 + c) * sbn + k0 + seg * 8);
            }
        } else {
            const unsigned short* B16 = (const unsigned short*)B;
            for (int i = t; i < 1024; i += 32) {
                int kk = i & 31, c = i >> 5;          // k fastest (contiguous)
                int k = k0 + kk, n = n0 + c;
                int kc = k < K ? k : K - 1;
                int nc = n < N ? n : N - 1;
                unsigned short raw = B16[(long long)nc * sbn + kc];
                raw &= (unsigned short)-(int)(k < K && n < N);
                *(unsigned short*)&Bs[c][kk] = raw;
            }
        }
        if (k0 + 32 < K)
            __builtin_prefetch((const void*)(A + (long long)m0 * lda + k0 + 32), 0, 3);
        __syncthreads();

        // fragment gather (ISA 7.12.2): two contiguous 8-half runs per operand
        bool hi = t >= 16;
        int  rw = t & 15;
        int  koa = hi ? 8 : 0;
        int  kob = hi ? 8 : 0;
        v8h a0 = *(const v8h*)&As[rw][koa];
        v8h a1 = *(const v8h*)&As[rw][16 + koa];
        v16h af = __builtin_shufflevector(a0, a1, 0,1,2,3,4,5,6,7,8,9,10,11,12,13,14,15);
        v8h b00 = *(const v8h*)&Bs[rw][kob];
        v8h b01 = *(const v8h*)&Bs[rw][kob + 16];
        v16h bf0 = __builtin_shufflevector(b00, b01, 0,1,2,3,4,5,6,7,8,9,10,11,12,13,14,15);
        v8h b10 = *(const v8h*)&Bs[16 + rw][kob];
        v8h b11 = *(const v8h*)&Bs[16 + rw][kob + 16];
        v16h bf1 = __builtin_shufflevector(b10, b11, 0,1,2,3,4,5,6,7,8,9,10,11,12,13,14,15);

        acc0 = __builtin_amdgcn_wmma_f32_16x16x32_f16(false, af, false, bf0,
                                                      (short)0, acc0, false, false);
        acc1 = __builtin_amdgcn_wmma_f32_16x16x32_f16(false, af, false, bf1,
                                                      (short)0, acc1, false, false);
        __syncthreads();
    }

    // store: C/D layout -- VGPR r: lanes 0-15 -> M=r, lanes 16-31 -> M=r+8
    bool hi = t >= 16;
    int  na = n0 + (t & 15);
    int  nb = na + 16;
    float bva = (bias && na < N) ? bias[na] : 0.f;
    float bvb = (bias && nb < N) ? bias[nb] : 0.f;
    #pragma unroll
    for (int r = 0; r < 8; ++r) {
        int m = m0 + r + (hi ? 8 : 0);
        if (m < M) {
            if (na < N) {
                float val = acc0[r] + bva;
                if (C) {
                    long long off = cOff + (long long)m * ldc + na;
                    C[off] = accumulate ? (C[off] + val) : val;
                }
                if (C16) C16[c16Off + (long long)m * ldc16 + na] = (_Float16)val;
            }
            if (nb < N) {
                float val = acc1[r] + bvb;
                if (C) {
                    long long off = cOff + (long long)m * ldc + nb;
                    C[off] = accumulate ? (C[off] + val) : val;
                }
                if (C16) C16[c16Off + (long long)m * ldc16 + nb] = (_Float16)val;
            }
        }
    }
}

// ---------------------------------------------------------------------------
// f32 -> f16 convert (grid stride)
// ---------------------------------------------------------------------------
__global__ void cvt_f32_f16(const float* __restrict__ in, _Float16* __restrict__ out,
                            long long n)
{
    long long i      = (long long)blockIdx.x * blockDim.x + threadIdx.x;
    long long stride = (long long)gridDim.x * blockDim.x;
    for (; i < n; i += stride) out[i] = (_Float16)in[i];
}

// Tiled f32 [K][N] -> f16 transposed [N][K], per z-slab (dim x dim square)
__global__ __launch_bounds__(256)
void cvt_t_f16(const float* __restrict__ in, _Float16* __restrict__ outT, int dim)
{
    __shared__ _Float16 tile[32][33];
    size_t slab = (size_t)blockIdx.z * dim * dim;
    const float* ip = in + slab;
    _Float16*    op = outT + slab;
    int x0 = blockIdx.x * 32, y0 = blockIdx.y * 32;
    int tx = threadIdx.x & 31, ty = threadIdx.x >> 5;      // 32 x 8
    #pragma unroll
    for (int j = 0; j < 32; j += 8)
        tile[ty + j][tx] = (_Float16)ip[(size_t)(y0 + ty + j) * dim + x0 + tx];
    __syncthreads();
    #pragma unroll
    for (int j = 0; j < 32; j += 8)
        op[(size_t)(x0 + ty + j) * dim + y0 + tx] = tile[tx][ty + j];
}

// Fold We (126 rows, dup-concat) -> WeF f16: wef[(lr)][64][512] (row 63 = 0)
// and transposed wefT[(lr)][512][64] for the ae @ WeF GEMM.
__global__ void fold_we(const float* __restrict__ We, _Float16* __restrict__ wef,
                        _Float16* __restrict__ wefT, long long total)
{
    long long i = (long long)blockIdx.x * blockDim.x + threadIdx.x;
    if (i >= total) return;
    int c = (int)(i & 511);
    long long t = i >> 9;
    int f = (int)(t & 63);
    long long lr = t >> 6;                 // 0..5
    long long l = lr / 3, r = lr % 3;
    float val = 0.f;
    if (f < PF_) {
        long long src = ((l * RR_ + r) * 126 + f) * 512 + c;
        val = We[src] + We[src + (long long)PF_ * 512];
    }
    _Float16 h = (_Float16)val;
    wef [(lr * PFP + f) * 512 + c] = h;
    wefT[(lr * 512 + c) * PFP + f] = h;
}

// NeRF positional encoding table pe[b, s, n, 63] (f16), delta = pos_dst[n]-pos_src[s]
__global__ void pe_kernel(const float* __restrict__ ps, const float* __restrict__ pd,
                          _Float16* __restrict__ pe, int Ns, int Nd, long long total)
{
    long long idx = (long long)blockIdx.x * blockDim.x + threadIdx.x;
    if (idx >= total) return;
    int n = (int)(idx % Nd);
    long long t = idx / Nd;
    int s = (int)(t % Ns);
    int b = (int)(t / Ns);
    float d[3];
    #pragma unroll
    for (int i = 0; i < 3; ++i)
        d[i] = pd[(b * Nd + n) * 3 + i] - ps[(b * Ns + s) * 3 + i];
    _Float16* o = pe + idx * (long long)PF_;
    o[0] = (_Float16)d[0]; o[1] = (_Float16)d[1]; o[2] = (_Float16)d[2];
    #pragma unroll
    for (int i = 0; i < 3; ++i) {
        #pragma unroll
        for (int j = 0; j < 10; ++j) {
            float ang = d[i] * (float)(1 << j);
            o[3 + i * 20 + j]      = (_Float16)__sinf(ang);
            o[3 + i * 20 + 10 + j] = (_Float16)__cosf(ang);
        }
    }
}

// ev[c] = edge_emb[0:126] . We[:, c]
__global__ void ev_kernel(const float* __restrict__ ee, const float* __restrict__ We,
                          float* __restrict__ ev)
{
    int c = blockIdx.x * blockDim.x + threadIdx.x;
    if (c >= HD_) return;
    float a = 0.f;
    for (int f = 0; f < 126; ++f) a += ee[f] * We[f * HD_ + c];
    ev[c] = a;
}

// out[b,n,c] += langv[b,c] + ev[c]
__global__ void bcast_add(float* __restrict__ o, const float* __restrict__ langv,
                          const float* __restrict__ ev, long long total, int NdH)
{
    long long i = (long long)blockIdx.x * blockDim.x + threadIdx.x;
    if (i >= total) return;
    int c = (int)(i & 511);
    int b = (int)(i / NdH);
    o[i] += langv[b * HD_ + c] + ev[c];
}

// ---------------------------------------------------------------------------
// Fused attention per (dst n, batch b): edge logits via qWe, per-head softmax,
// alpha.v accumulation into outAcc, ae[f,h] = sum_s alpha*pe (stride-64 pad).
// ---------------------------------------------------------------------------
__global__ __launch_bounds__(256)
void attn_kernel(const float* __restrict__ scores,   // [B, 8, Nd, Ns]
                 const float* __restrict__ qWe,      // [8, BNd, 64]
                 const _Float16* __restrict__ pe,    // [B, Ns, Nd, 63]
                 const float* __restrict__ v,        // [B, Ns, 512]
                 float* __restrict__ outAcc,         // [B, Nd, 512] (+=)
                 _Float16* __restrict__ ae,          // [8, BNd, 64]
                 int Ns, int Nd, int BNd)
{
    __shared__ float qwes[PF_ * NH_];
    __shared__ float la[256 * NH_];
    __shared__ float hinv[NH_];

    int n = blockIdx.x, b = blockIdx.y, tid = threadIdx.x;
    int bn = b * Nd + n;

    for (int i = tid; i < PF_ * NH_; i += 256) {
        int f = i >> 3, h = i & 7;
        qwes[f * NH_ + h] = qWe[((long long)h * BNd + bn) * PFP + f];
    }
    __syncthreads();

    int s = tid;
    if (s < Ns) {
        const _Float16* pep = pe + ((long long)(b * Ns + s) * Nd + n) * PF_;
        float qe[NH_] = {0.f, 0.f, 0.f, 0.f, 0.f, 0.f, 0.f, 0.f};
        for (int f = 0; f < PF_; ++f) {
            float p = (float)pep[f];
            #pragma unroll
            for (int h = 0; h < NH_; ++h) qe[h] += p * qwes[f * NH_ + h];
        }
        const float* sc = scores + ((long long)b * NH_ * Nd + n) * Ns + s;
        #pragma unroll
        for (int h = 0; h < NH_; ++h)
            la[s * NH_ + h] = (sc[(long long)h * Nd * Ns] + qe[h]) * 0.125f;
    }
    __syncthreads();

    // per-head softmax: 8 heads x 32 lanes
    {
        int h = tid >> 5, lane = tid & 31;
        float m = -3.0e38f;
        for (int ss = lane; ss < Ns; ss += 32) m = fmaxf(m, la[ss * NH_ + h]);
        for (int off = 16; off; off >>= 1) m = fmaxf(m, __shfl_xor(m, off, 32));
        float sum = 0.f;
        for (int ss = lane; ss < Ns; ss += 32) {
            float e = __expf(la[ss * NH_ + h] - m);
            la[ss * NH_ + h] = e;
            sum += e;
        }
        for (int off = 16; off; off >>= 1) sum += __shfl_xor(sum, off, 32);
        if (lane == 0) hinv[h] = 1.f / sum;
    }
    __syncthreads();

    // out_v: each thread 2 columns
    #pragma unroll
    for (int rep = 0; rep < 2; ++rep) {
        int c  = tid + rep * 256;
        int hh = c >> 6;
        float accv = 0.f;
        const float* vp = v + (long long)b * Ns * HD_ + c;
        for (int ss = 0; ss < Ns; ++ss)
            accv += la[ss * NH_ + hh] * vp[(long long)ss * HD_];
        outAcc[(long long)bn * HD_ + c] += accv * hinv[hh];
    }

    // ae[f,h] = sum_s alpha * pe (f = 63 written as zero pad)
    for (int i = tid; i < PFP * NH_; i += 256) {
        int f = i >> 3, hh = i & 7;
        float res = 0.f;
        if (f < PF_) {
            float acca = 0.f;
            const _Float16* pep = pe + ((long long)b * Ns * Nd + n) * PF_ + f;
            for (int ss = 0; ss < Ns; ++ss)
                acca += la[ss * NH_ + hh] * (float)pep[(long long)ss * Nd * PF_];
            res = acca * hinv[hh];
        }
        ae[((long long)hh * BNd + bn) * PFP + f] = (_Float16)res;
    }
}

// ---------------------------------------------------------------------------
// Fused ReLU + LayerNorm (row = 512), writes f32 and f16 copies
// ---------------------------------------------------------------------------
__global__ __launch_bounds__(256)
void ln_relu(const float* __restrict__ x, const float* __restrict__ g,
             const float* __restrict__ bt, float* __restrict__ y,
             _Float16* __restrict__ y16)
{
    __shared__ float s1[8], s2[8];
    long long row = blockIdx.x;
    int tid = threadIdx.x;
    float v0 = fmaxf(x[row * HD_ + tid], 0.f);
    float v1 = fmaxf(x[row * HD_ + tid + 256], 0.f);
    float a = v0 + v1, q = v0 * v0 + v1 * v1;
    for (int off = 16; off; off >>= 1) {
        a += __shfl_xor(a, off, 32);
        q += __shfl_xor(q, off, 32);
    }
    if ((tid & 31) == 0) { s1[tid >> 5] = a; s2[tid >> 5] = q; }
    __syncthreads();
    float tot = 0.f, tot2 = 0.f;
    #pragma unroll
    for (int i = 0; i < 8; ++i) { tot += s1[i]; tot2 += s2[i]; }
    float mean = tot * (1.f / HD_);
    float var  = tot2 * (1.f / HD_) - mean * mean;
    float rs   = rsqrtf(var + 1e-5f);
    float o0 = (v0 - mean) * rs * g[tid]       + bt[tid];
    float o1 = (v1 - mean) * rs * g[tid + 256] + bt[tid + 256];
    y[row * HD_ + tid]        = o0;
    y[row * HD_ + tid + 256]  = o1;
    y16[row * HD_ + tid]       = (_Float16)o0;
    y16[row * HD_ + tid + 256] = (_Float16)o1;
}

// ---------------------------------------------------------------------------
extern "C" void kernel_launch(void* const* d_in, const int* in_sizes, int n_in,
                              void* d_out, int out_size, void* d_ws, size_t ws_size,
                              hipStream_t stream)
{
    (void)in_sizes; (void)n_in; (void)out_size; (void)ws_size;
    const float* scene_x     = (const float*)d_in[0];
    const float* scene_pos   = (const float*)d_in[1];
    const float* gripper_x   = (const float*)d_in[2];
    const float* gripper_pos = (const float*)d_in[3];
    const float* lang_emb    = (const float*)d_in[4];
    const float* Wlp         = (const float*)d_in[5];
    const float* blp         = (const float*)d_in[6];
    const float* edge_emb    = (const float*)d_in[7];
    const float* Wq          = (const float*)d_in[8];
    const float* bq          = (const float*)d_in[9];
    const float* Wk          = (const float*)d_in[10];
    const float* bk          = (const float*)d_in[11];
    const float* Wv          = (const float*)d_in[12];
    const float* bv          = (const float*)d_in[13];
    const float* We          = (const float*)d_in[14];
    const float* Wsk         = (const float*)d_in[15];
    const float* bsk         = (const float*)d_in[16];
    const float* lng         = (const float*)d_in[17];
    const float* lnb         = (const float*)d_in[18];
    float* out = (float*)d_out;

    const int BNS = BN_ * NS_;   // 1024
    const int BNG = BN_ * NG_;   // 64

    char* wp = (char*)d_ws;
    auto alloc = [&](size_t bytes) -> char* {
        char* r = wp; wp += (bytes + 255) & ~(size_t)255; return r;
    };
    // all weights stored TRANSPOSED: [out(n)][in(k)], k contiguous
    _Float16* wqT   = (_Float16*)alloc(sizeof(_Float16) * (size_t)LL_ * RR_ * HD_ * HD_);
    _Float16* wkT   = (_Float16*)alloc(sizeof(_Float16) * (size_t)LL_ * RR_ * HD_ * HD_);
    _Float16* wvT   = (_Float16*)alloc(sizeof(_Float16) * (size_t)LL_ * RR_ * HD_ * HD_);
    _Float16* wsT   = (_Float16*)alloc(sizeof(_Float16) * (size_t)LL_ * RR_ * HD_ * HD_);
    _Float16* wlpT  = (_Float16*)alloc(sizeof(_Float16) * (size_t)HD_ * HD_);
    _Float16* wef   = (_Float16*)alloc(sizeof(_Float16) * (size_t)LL_ * 3 * PFP * HD_);
    _Float16* wefT  = (_Float16*)alloc(sizeof(_Float16) * (size_t)LL_ * 3 * HD_ * PFP);
    _Float16* sx16  = (_Float16*)alloc(sizeof(_Float16) * (size_t)BNS * HD_);
    _Float16* gx16  = (_Float16*)alloc(sizeof(_Float16) * (size_t)BNG * HD_);
    _Float16* le16  = (_Float16*)alloc(sizeof(_Float16) * (size_t)BN_ * HD_);
    _Float16* lang16= (_Float16*)alloc(sizeof(_Float16) * (size_t)BN_ * HD_);
    _Float16* pe_ss = (_Float16*)alloc(sizeof(_Float16) * (size_t)BN_ * NS_ * NS_ * PF_);
    _Float16* pe_sg = (_Float16*)alloc(sizeof(_Float16) * (size_t)BN_ * NS_ * NG_ * PF_);
    _Float16* pe_gg = (_Float16*)alloc(sizeof(_Float16) * (size_t)BN_ * NG_ * NG_ * PF_);
    _Float16* q16   = (_Float16*)alloc(sizeof(_Float16) * (size_t)BNS * HD_);
    _Float16* k16   = (_Float16*)alloc(sizeof(_Float16) * (size_t)BNS * HD_);
    float*    vbuf  = (float*)alloc(sizeof(float) * (size_t)BNS * HD_);
    float*    scoreb= (float*)alloc(sizeof(float) * (size_t)BN_ * NH_ * NS_ * NS_);
    float*    qwe   = (float*)alloc(sizeof(float) * (size_t)NH_ * BNS * PFP);
    _Float16* ae16  = (_Float16*)alloc(sizeof(_Float16) * (size_t)NH_ * BNS * PFP);
    float*    snew  = (float*)alloc(sizeof(float) * (size_t)BNS * HD_);
    float*    gnew  = (float*)alloc(sizeof(float) * (size_t)BNG * HD_);
    float*    scur  = (float*)alloc(sizeof(float) * (size_t)BNS * HD_);
    float*    gcur  = (float*)alloc(sizeof(float) * (size_t)BNG * HD_);
    float*    langv = (float*)alloc(sizeof(float) * (size_t)BN_ * HD_);
    float*    evb   = (float*)alloc(sizeof(float) * (size_t)HD_);

    auto cvt = [&](const float* in, _Float16* o, long long n) {
        long long blocks = (n + 255) / 256; if (blocks > 8192) blocks = 8192;
        cvt_f32_f16<<<dim3((unsigned)blocks), 256, 0, stream>>>(in, o, n);
    };
    auto gemm = [&](const _Float16* A, int lda, long long sA1, long long sA2,
                    const _Float16* B, int sbn, long long sB1, long long sB2,
                    float* C, int ldc, long long sC1, long long sC2,
                    _Float16* C16, int ldc16, long long sC161, long long sC162,
                    const float* bias, int M, int N, int K,
                    int batches, int innerB, int accum) {
        dim3 grid((N + 31) / 32, (M + 15) / 16, batches);
        gemm_wmma<<<grid, 32, 0, stream>>>(A, lda, sA1, sA2, B, sbn, sB1, sB2,
                                           C, ldc, sC1, sC2, C16, ldc16, sC161, sC162,
                                           bias, M, N, K, innerB, accum);
    };

    // ---- one-time conversions / tables ----
    cvt(scene_x,   sx16, (long long)BNS * HD_);
    cvt(gripper_x, gx16, (long long)BNG * HD_);
    cvt(lang_emb,  le16, (long long)BN_ * HD_);
    {
        dim3 tg(HD_ / 32, HD_ / 32, LL_ * RR_);
        cvt_t_f16<<<tg, 256, 0, stream>>>(Wq,  wqT, HD_);
        cvt_t_f16<<<tg, 256, 0, stream>>>(Wk,  wkT, HD_);
        cvt_t_f16<<<tg, 256, 0, stream>>>(Wv,  wvT, HD_);
        cvt_t_f16<<<tg, 256, 0, stream>>>(Wsk, wsT, HD_);
        cvt_t_f16<<<dim3(HD_ / 32, HD_ / 32, 1), 256, 0, stream>>>(Wlp, wlpT, HD_);
    }
    {
        long long tot = (long long)LL_ * 3 * PFP * HD_;
        fold_we<<<dim3((unsigned)((tot + 255) / 256)), 256, 0, stream>>>(We, wef, wefT, tot);
    }
    {
        long long t1 = (long long)BN_ * NS_ * NS_;
        pe_kernel<<<dim3((unsigned)((t1 + 255) / 256)), 256, 0, stream>>>(scene_pos, scene_pos, pe_ss, NS_, NS_, t1);
        long long t2 = (long long)BN_ * NS_ * NG_;
        pe_kernel<<<dim3((unsigned)((t2 + 255) / 256)), 256, 0, stream>>>(scene_pos, gripper_pos, pe_sg, NS_, NG_, t2);
        long long t3 = (long long)BN_ * NG_ * NG_;
        pe_kernel<<<dim3((unsigned)((t3 + 255) / 256)), 256, 0, stream>>>(gripper_pos, gripper_pos, pe_gg, NG_, NG_, t3);
    }
    // lang projection -> f16
    gemm(le16, HD_, 0, 0, wlpT, HD_, 0, 0, nullptr, 0, 0, 0,
         lang16, HD_, 0, 0, blp, BN_, HD_, HD_, 1, 1, 0);

    for (int l = 0; l < LL_; ++l) {
        auto WQT = [&](int r) { return wqT + (size_t)(l * RR_ + r) * HD_ * HD_; };
        auto WKT = [&](int r) { return wkT + (size_t)(l * RR_ + r) * HD_ * HD_; };
        auto WVT = [&](int r) { return wvT + (size_t)(l * RR_ + r) * HD_ * HD_; };
        auto WST = [&](int r) { return wsT + (size_t)(l * RR_ + r) * HD_ * HD_; };
        auto WEF = [&](int r) { return wef  + (size_t)(l * 3 + r) * PFP * HD_; };
        auto WEFT= [&](int r) { return wefT + (size_t)(l * 3 + r) * HD_ * PFP; };
        auto BQ  = [&](int r) { return bq  + (size_t)(l * RR_ + r) * HD_; };
        auto BK  = [&](int r) { return bk  + (size_t)(l * RR_ + r) * HD_; };
        auto BV  = [&](int r) { return bv  + (size_t)(l * RR_ + r) * HD_; };
        auto BS  = [&](int r) { return bsk + (size_t)(l * RR_ + r) * HD_; };
        auto WER = [&](int r) { return We  + (size_t)(l * RR_ + r) * 126 * HD_; };

        // ===================== scene update: r=0 dense + r=3 lang =====================
        gemm(sx16, HD_, 0, 0, WQT(0), HD_, 0, 0, nullptr, 0, 0, 0,
             q16, HD_, 0, 0, BQ(0), BNS, HD_, HD_, 1, 1, 0);
        gemm(sx16, HD_, 0, 0, WKT(0), HD_, 0, 0, nullptr, 0, 0, 0,
             k16, HD_, 0, 0, BK(0), BNS, HD_, HD_, 1, 1, 0);
        gemm(sx16, HD_, 0, 0, WVT(0), HD_, 0, 0, vbuf, HD_, 0, 0,
             nullptr, 0, 0, 0, BV(0), BNS, HD_, HD_, 1, 1, 0);
        gemm(sx16, HD_, 0, 0, WST(0), HD_, 0, 0, snew, HD_, 0, 0,
             nullptr, 0, 0, 0, BS(0), BNS, HD_, HD_, 1, 1, 0);       // beta = 0
        gemm(sx16, HD_, 0, 0, WST(3), HD_, 0, 0, snew, HD_, 0, 0,
             nullptr, 0, 0, 0, BS(3), BNS, HD_, HD_, 1, 1, 1);       // += lang skip
        gemm(lang16, HD_, 0, 0, WVT(3), HD_, 0, 0, langv, HD_, 0, 0,
             nullptr, 0, 0, 0, BV(3), BN_, HD_, HD_, 1, 1, 0);
        ev_kernel<<<2, 256, 0, stream>>>(edge_emb, WER(3), evb);
        bcast_add<<<dim3((BNS * HD_) / 256), 256, 0, stream>>>(snew, langv, evb,
                                                               (long long)BNS * HD_, NS_ * HD_);
        // scores[b,h,n,s] = q . k
        gemm(q16, HD_, (long long)NS_ * HD_, 64, k16, HD_, (long long)NS_ * HD_, 64,
             scoreb, NS_, 8LL * NS_ * NS_, (long long)NS_ * NS_,
             nullptr, 0, 0, 0, nullptr, NS_, NS_, 64, BN_ * NH_, NH_, 0);
        // qWe[h, bn, f] (f padded to 64)
        gemm(q16, HD_, 0, 64, WEF(0), HD_, 0, 64,
             qwe, PFP, 0, (long long)BNS * PFP,
             nullptr, 0, 0, 0, nullptr, BNS, PFP, 64, NH_, NH_, 0);
        attn_kernel<<<dim3(NS_, BN_), 256, 0, stream>>>(scoreb, qwe, pe_ss, vbuf,
                                                        snew, ae16, NS_, NS_, BNS);
        gemm(ae16, PFP, 0, (long long)BNS * PFP, WEFT(0), PFP, 0, 64LL * PFP,
             snew, HD_, 0, 64, nullptr, 0, 0, 0, nullptr, BNS, 64, PFP, NH_, NH_, 1);

        // ===================== gripper update: r=1 (s->g) =====================
        gemm(gx16, HD_, 0, 0, WQT(1), HD_, 0, 0, nullptr, 0, 0, 0,
             q16, HD_, 0, 0, BQ(1), BNG, HD_, HD_, 1, 1, 0);
        gemm(sx16, HD_, 0, 0, WKT(1), HD_, 0, 0, nullptr, 0, 0, 0,
             k16, HD_, 0, 0, BK(1), BNS, HD_, HD_, 1, 1, 0);
        gemm(sx16, HD_, 0, 0, WVT(1), HD_, 0, 0, vbuf, HD_, 0, 0,
             nullptr, 0, 0, 0, BV(1), BNS, HD_, HD_, 1, 1, 0);
        gemm(gx16, HD_, 0, 0, WST(1), HD_, 0, 0, gnew, HD_, 0, 0,
             nullptr, 0, 0, 0, BS(1), BNG, HD_, HD_, 1, 1, 0);       // beta = 0
        gemm(q16, HD_, (long long)NG_ * HD_, 64, k16, HD_, (long long)NS_ * HD_, 64,
             scoreb, NS_, 8LL * NG_ * NS_, (long long)NG_ * NS_,
             nullptr, 0, 0, 0, nullptr, NG_, NS_, 64, BN_ * NH_, NH_, 0);
        gemm(q16, HD_, 0, 64, WEF(1), HD_, 0, 64,
             qwe, PFP, 0, (long long)BNG * PFP,
             nullptr, 0, 0, 0, nullptr, BNG, PFP, 64, NH_, NH_, 0);
        attn_kernel<<<dim3(NG_, BN_), 256, 0, stream>>>(scoreb, qwe, pe_sg, vbuf,
                                                        gnew, ae16, NS_, NG_, BNG);
        gemm(ae16, PFP, 0, (long long)BNG * PFP, WEFT(1), PFP, 0, 64LL * PFP,
             gnew, HD_, 0, 64, nullptr, 0, 0, 0, nullptr, BNG, 64, PFP, NH_, NH_, 1);

        // ===================== gripper: r=2 (g->g) =====================
        gemm(gx16, HD_, 0, 0, WQT(2), HD_, 0, 0, nullptr, 0, 0, 0,
             q16, HD_, 0, 0, BQ(2), BNG, HD_, HD_, 1, 1, 0);
        gemm(gx16, HD_, 0, 0, WKT(2), HD_, 0, 0, nullptr, 0, 0, 0,
             k16, HD_, 0, 0, BK(2), BNG, HD_, HD_, 1, 1, 0);
        gemm(gx16, HD_, 0, 0, WVT(2), HD_, 0, 0, vbuf, HD_, 0, 0,
             nullptr, 0, 0, 0, BV(2), BNG, HD_, HD_, 1, 1, 0);
        gemm(gx16, HD_, 0, 0, WST(2), HD_, 0, 0, gnew, HD_, 0, 0,
             nullptr, 0, 0, 0, BS(2), BNG, HD_, HD_, 1, 1, 1);       // +=
        gemm(q16, HD_, (long long)NG_ * HD_, 64, k16, HD_, (long long)NG_ * HD_, 64,
             scoreb, NG_, 8LL * NG_ * NG_, (long long)NG_ * NG_,
             nullptr, 0, 0, 0, nullptr, NG_, NG_, 64, BN_ * NH_, NH_, 0);
        gemm(q16, HD_, 0, 64, WEF(2), HD_, 0, 64,
             qwe, PFP, 0, (long long)BNG * PFP,
             nullptr, 0, 0, 0, nullptr, BNG, PFP, 64, NH_, NH_, 0);
        attn_kernel<<<dim3(NG_, BN_), 256, 0, stream>>>(scoreb, qwe, pe_gg, vbuf,
                                                        gnew, ae16, NG_, NG_, BNG);
        gemm(ae16, PFP, 0, (long long)BNG * PFP, WEFT(2), PFP, 0, 64LL * PFP,
             gnew, HD_, 0, 64, nullptr, 0, 0, 0, nullptr, BNG, 64, PFP, NH_, NH_, 1);

        // ===================== gripper: r=4 lang =====================
        gemm(gx16, HD_, 0, 0, WST(4), HD_, 0, 0, gnew, HD_, 0, 0,
             nullptr, 0, 0, 0, BS(4), BNG, HD_, HD_, 1, 1, 1);
        gemm(lang16, HD_, 0, 0, WVT(4), HD_, 0, 0, langv, HD_, 0, 0,
             nullptr, 0, 0, 0, BV(4), BN_, HD_, HD_, 1, 1, 0);
        ev_kernel<<<2, 256, 0, stream>>>(edge_emb, WER(4), evb);
        bcast_add<<<dim3((BNG * HD_) / 256), 256, 0, stream>>>(gnew, langv, evb,
                                                               (long long)BNG * HD_, NG_ * HD_);

        // ===================== ReLU + LayerNorm =====================
        ln_relu<<<dim3(BNS), 256, 0, stream>>>(snew, lng + (l * 2 + 0) * HD_,
                                               lnb + (l * 2 + 0) * HD_, scur, sx16);
        ln_relu<<<dim3(BNG), 256, 0, stream>>>(gnew, lng + (l * 2 + 1) * HD_,
                                               lnb + (l * 2 + 1) * HD_,
                                               (l == LL_ - 1) ? out : gcur, gx16);
    }
}
